// Block_20469814132941
// MI455X (gfx1250) — matrix operand order repaired and checked by reading, not compiled, not run on hardware
//
#include <hip/hip_runtime.h>
#include <hip/hip_bf16.h>
#include <cstdint>

typedef _Float16 h16;
typedef __attribute__((ext_vector_type(16))) _Float16 h16x16;
typedef __attribute__((ext_vector_type(8)))  _Float16 h16x8;
typedef __attribute__((ext_vector_type(8)))  float    f32x8;
typedef __attribute__((ext_vector_type(4)))  uint32_t u32x4;

// ---------------------------------------------------------------------------
// WMMA fragment loaders (wave32), contiguous k order in A and B, not the ISA table order.
// A 16x32 f16: lanes 0-15 / 16-31 both hold rows M=0..15; VGPR i -> K=2i,2i+1
// (+16 for hi lanes), matching the contiguous B layout.
// ---------------------------------------------------------------------------
__device__ __forceinline__ h16x16 load_frag_a(const h16* base, int stride) {
  const int lane = threadIdx.x & 31;
  const h16* p = base + (size_t)(lane & 15) * stride + ((lane >> 4) & 1) * 16;
  const uint32_t* q = (const uint32_t*)p;
  union { h16x16 v; uint32_t u[8]; } r;
  r.u[0] = q[0]; r.u[1] = q[1]; r.u[2] = q[2]; r.u[3] = q[3];
  r.u[4] = q[4]; r.u[5] = q[5]; r.u[6] = q[6]; r.u[7] = q[7];
  return r.v;
}

// B 32x16 f16: lane = column n (lane&15); lanes 0-15 hold K=0..15 contiguous,
// lanes 16-31 hold K=16..31. Requires B stored column-major (row n = column).
__device__ __forceinline__ h16x16 load_frag_b(const h16* base, int stride) {
  const int lane = threadIdx.x & 31;
  const h16* p = base + (size_t)(lane & 15) * stride + ((lane >> 4) & 1) * 16;
  const u32x4* q = (const u32x4*)p;
  union { h16x16 v; u32x4 u[2]; } r;
  r.u[0] = q[0]; r.u[1] = q[1];
  return r.v;
}

__device__ __forceinline__ f32x8 zero8() {
  f32x8 z;
  #pragma unroll
  for (int i = 0; i < 8; i++) z[i] = 0.0f;
  return z;
}

__device__ __forceinline__ float gelu_f(float x) {
  const float c = 0.7978845608028654f; // sqrt(2/pi)
  return 0.5f * x * (1.0f + tanhf(c * (x + 0.044715f * x * x * x)));
}

// ---------------------------------------------------------------------------
// CDNA5 async global->LDS copy (16B per lane, ASYNCcnt-tracked).
// LDS operand = wave-relative LDS byte address; generic pointers to __shared__
// carry that offset in their low 32 bits (ISA aperture rules).
// ---------------------------------------------------------------------------
__device__ __forceinline__ void async_copy16(const h16* g, h16* l) {
  const uint32_t loff = (uint32_t)(uintptr_t)l;
  asm volatile("global_load_async_to_lds_b128 %0, %1, off"
               :: "v"(loff), "v"(g)
               : "memory");
}

__device__ __forceinline__ void wait_async0() {
  asm volatile("s_wait_asynccnt 0" ::: "memory");
}

// ---------------------------------------------------------------------------
// Elementwise f32 -> f16 cast
// ---------------------------------------------------------------------------
__global__ void cast_f32_f16(const float* __restrict__ in, h16* __restrict__ out, int n) {
  int i = blockIdx.x * blockDim.x + threadIdx.x;
  if (i < n) out[i] = (h16)in[i];
}

// ---------------------------------------------------------------------------
// Cast + transpose: in (K x N, f32, row-major) -> out (N x K, f16, row-major)
// ---------------------------------------------------------------------------
__global__ __launch_bounds__(256) void transpose_cast(
    const float* __restrict__ in, h16* __restrict__ out, int K, int N) {
  __shared__ float t[32][33];
  const int n0 = blockIdx.x * 32, k0 = blockIdx.y * 32;
  const int tx = threadIdx.x & 31, ty = threadIdx.x >> 5;
  #pragma unroll
  for (int j = 0; j < 32; j += 8)
    t[ty + j][tx] = in[(size_t)(k0 + ty + j) * N + n0 + tx];
  __syncthreads();
  #pragma unroll
  for (int j = 0; j < 32; j += 8)
    out[(size_t)(n0 + ty + j) * K + k0 + tx] = (h16)t[tx][ty + j];
}

// ---------------------------------------------------------------------------
// Tiled WMMA GEMM: C = A(MxK,f16) * Bt(NxK,f16 transposed weights) + bias
// Tile 128x128, K-step 32; 8 waves -> each wave 32x64 (2x4 WMMA frags).
// Double-buffered LDS, async global->LDS staging overlapped with WMMA.
// Epilogue: optional residual (res!=null), optional GELU (gelu=1),
// writes f32 (outF) and/or f16 (outH).
// ---------------------------------------------------------------------------
#define LDSW 40   // padded LDS row stride (halves): 80B, 16B-aligned, conflict-free

__device__ __forceinline__ void stage_async(
    const h16* __restrict__ A, const h16* __restrict__ Bt,
    h16* sA, h16* sB, int tid, int rbase, int cbase, int K, int k0) {
  #pragma unroll
  for (int i = 0; i < 2; i++) {
    int c = tid * 2 + i;          // 0..511 chunks of 8 halves
    int r = c >> 2;               // 0..127
    int cg = c & 3;               // 0..3
    async_copy16(&A[(size_t)(rbase + r) * K + k0 + cg * 8], &sA[r * LDSW + cg * 8]);
    async_copy16(&Bt[(size_t)(cbase + r) * K + k0 + cg * 8], &sB[r * LDSW + cg * 8]);
  }
}

__global__ __launch_bounds__(256) void gemm_f16(
    const h16* __restrict__ A, const h16* __restrict__ Bt,
    const float* __restrict__ bias, const float* __restrict__ res,
    float* __restrict__ outF, h16* __restrict__ outH,
    int M, int N, int K, int gelu) {
  __shared__ h16 sA[2][128 * LDSW];
  __shared__ h16 sB[2][128 * LDSW];
  const int tid = threadIdx.x;
  const int lane = tid & 31, wave = tid >> 5;
  const int wm = wave >> 1, wn = wave & 1;
  const int rbase = blockIdx.y * 128;
  const int cbase = blockIdx.x * 128;

  f32x8 acc[2][4];
  #pragma unroll
  for (int i = 0; i < 2; i++)
    #pragma unroll
    for (int j = 0; j < 4; j++) acc[i][j] = zero8();

  const int nk = K >> 5;
  stage_async(A, Bt, sA[0], sB[0], tid, rbase, cbase, K, 0);

  for (int t = 0; t < nk; t++) {
    const int cur = t & 1;
    wait_async0();        // own async loads for buf `cur` complete
    __syncthreads();      // everyone's loads complete; prev compute on other buf done
    if (t + 1 < nk)
      stage_async(A, Bt, sA[cur ^ 1], sB[cur ^ 1], tid, rbase, cbase, K, (t + 1) * 32);

    const h16* cA = sA[cur];
    const h16* cB = sB[cur];
    h16x16 fa[2], fb[4];
    #pragma unroll
    for (int i = 0; i < 2; i++)
      fa[i] = load_frag_a(&cA[(wm * 32 + i * 16) * LDSW], LDSW);
    #pragma unroll
    for (int j = 0; j < 4; j++)
      fb[j] = load_frag_b(&cB[(wn * 64 + j * 16) * LDSW], LDSW);

    #pragma unroll
    for (int i = 0; i < 2; i++)
      #pragma unroll
      for (int j = 0; j < 4; j++)
        acc[i][j] = __builtin_amdgcn_wmma_f32_16x16x32_f16(
            false, fa[i], false, fb[j], (short)0, acc[i][j], false, false);
  }

  // epilogue: C/D layout -> col n = lane&15, row = vgpr + 8*(lane>>4)
  const int n = lane & 15, hi = (lane >> 4) & 1;
  #pragma unroll
  for (int i = 0; i < 2; i++) {
    #pragma unroll
    for (int j = 0; j < 4; j++) {
      const int c = cbase + wn * 64 + j * 16 + n;
      const float bv = bias[c];
      #pragma unroll
      for (int vi = 0; vi < 8; vi++) {
        const int r = rbase + wm * 32 + i * 16 + hi * 8 + vi;
        const size_t idx = (size_t)r * N + c;
        float v = acc[i][j][vi] + bv;
        if (res) v += res[idx];
        if (gelu) v = gelu_f(v);
        if (outF) outF[idx] = v;
        if (outH) outH[idx] = (h16)v;
      }
    }
  }
}

// ---------------------------------------------------------------------------
// Flash attention (causal). qkv: [4096][3072] f16 (q | k | v per 1024 cols).
// Block = one 128-row query tile of one (b,h); 8 waves x 16 q-rows each.
// K tile staged via async global->LDS; V transposed through VGPRs.
// y: [4096][1024] f16 (pre-Wo attention output).
// ---------------------------------------------------------------------------
__global__ __launch_bounds__(256) void attn_kernel(
    const h16* __restrict__ qkv, h16* __restrict__ y) {
  __shared__ h16 sK[64 * 72];        // [key][d]   (d-contiguous -> B frags for QK^T)
  __shared__ h16 sV[64 * 72];        // [d][key]   (key-contiguous -> B frags for PV)
  __shared__ h16 sP[8 * 16 * 72];    // per-wave P staging (16 rows x 64 keys)

  const int tid = threadIdx.x, lane = tid & 31, wave = tid >> 5;
  const int qt = blockIdx.x, h = blockIdx.y, b = blockIdx.z;
  const int n = lane & 15, hi = (lane >> 4) & 1;
  const size_t rowB = (size_t)b * 2048;
  const int qw = qt * 128 + wave * 16;   // wave's first query row (local T index)

  // Q fragments straight from global (A layout gather, d0..31 and d32..63)
  h16x16 qa[2];
  #pragma unroll
  for (int s = 0; s < 2; s++)
    qa[s] = load_frag_a(&qkv[(rowB + qw) * 3072 + h * 64 + s * 32], 3072);

  f32x8 o[4];
  #pragma unroll
  for (int dc = 0; dc < 4; dc++) o[dc] = zero8();
  float mi[8], li[8];
  #pragma unroll
  for (int vi = 0; vi < 8; vi++) { mi[vi] = -__builtin_inff(); li[vi] = 0.0f; }

  const int ntiles = qt * 2 + 2;   // key tiles of 64 covering causal span
  for (int j = 0; j < ntiles; j++) {
    const int kb = j * 64;
    __syncthreads();
    // cooperative stage: K tile async row-major, V tile transposed via VGPRs
    #pragma unroll
    for (int i = 0; i < 2; i++) {
      int c = tid * 2 + i;   // 0..511
      int r = c >> 3;        // key row 0..63
      int cg = c & 7;        // 8-half chunk of d
      size_t grow = (rowB + kb + r) * 3072;
      async_copy16(&qkv[grow + 1024 + h * 64 + cg * 8], &sK[r * 72 + cg * 8]);
      h16x8 vv = *(const h16x8*)&qkv[grow + 2048 + h * 64 + cg * 8];
      #pragma unroll
      for (int t = 0; t < 8; t++) sV[(cg * 8 + t) * 72 + r] = vv[t];
    }
    wait_async0();
    __syncthreads();

    // S = Q K^T  (16 q x 64 keys per wave)
    f32x8 sf[4];
    #pragma unroll
    for (int nc = 0; nc < 4; nc++) {
      f32x8 s = zero8();
      #pragma unroll
      for (int ss = 0; ss < 2; ss++)
        s = __builtin_amdgcn_wmma_f32_16x16x32_f16(
            false, qa[ss], false,
            load_frag_b(&sK[nc * 16 * 72 + ss * 32], 72),
            (short)0, s, false, false);
      sf[nc] = s;
    }
    // scale + causal mask
    #pragma unroll
    for (int nc = 0; nc < 4; nc++) {
      const int kc = kb + nc * 16 + n;
      #pragma unroll
      for (int vi = 0; vi < 8; vi++) {
        const int qr = qw + hi * 8 + vi;
        const float v = sf[nc][vi] * 0.125f;   // 1/sqrt(64)
        sf[nc][vi] = (kc <= qr) ? v : -__builtin_inff();
      }
    }
    // online softmax (rows live on vgpr index; reduce across 16 lanes)
    #pragma unroll
    for (int vi = 0; vi < 8; vi++) {
      float mx = fmaxf(fmaxf(sf[0][vi], sf[1][vi]), fmaxf(sf[2][vi], sf[3][vi]));
      #pragma unroll
      for (int off = 8; off >= 1; off >>= 1)
        mx = fmaxf(mx, __shfl_xor(mx, off, 32));
      const float mnew = fmaxf(mi[vi], mx);
      const float fac = __expf(mi[vi] - mnew);
      float rs = 0.0f;
      #pragma unroll
      for (int nc = 0; nc < 4; nc++) {
        const float e = __expf(sf[nc][vi] - mnew);
        sf[nc][vi] = e;
        rs += e;
      }
      #pragma unroll
      for (int off = 8; off >= 1; off >>= 1)
        rs += __shfl_xor(rs, off, 32);
      li[vi] = li[vi] * fac + rs;
      mi[vi] = mnew;
      #pragma unroll
      for (int dc = 0; dc < 4; dc++) o[dc][vi] *= fac;
    }
    // stage P (f16) into per-wave LDS region, then re-read as A fragments
    h16* pw = &sP[wave * 16 * 72];
    #pragma unroll
    for (int nc = 0; nc < 4; nc++)
      #pragma unroll
      for (int vi = 0; vi < 8; vi++)
        pw[(hi * 8 + vi) * 72 + nc * 16 + n] = (h16)sf[nc][vi];
    __syncthreads();

    // O += P V
    #pragma unroll
    for (int dc = 0; dc < 4; dc++)
      #pragma unroll
      for (int ss = 0; ss < 2; ss++)
        o[dc] = __builtin_amdgcn_wmma_f32_16x16x32_f16(
            false, load_frag_a(&pw[ss * 32], 72), false,
            load_frag_b(&sV[dc * 16 * 72 + ss * 32], 72),
            (short)0, o[dc], false, false);
  }

  // normalize and store y (f16)
  #pragma unroll
  for (int dc = 0; dc < 4; dc++)
    #pragma unroll
    for (int vi = 0; vi < 8; vi++) {
      const size_t r = rowB + qw + hi * 8 + vi;
      y[r * 1024 + h * 64 + dc * 16 + n] = (h16)(o[dc][vi] / li[vi]);
    }
}

// ---------------------------------------------------------------------------
// Host-side orchestration
// ---------------------------------------------------------------------------
extern "C" void kernel_launch(void* const* d_in, const int* in_sizes, int n_in,
                              void* d_out, int out_size, void* d_ws, size_t ws_size,
                              hipStream_t stream) {
  (void)in_sizes; (void)n_in; (void)out_size; (void)ws_size;
  const float* x     = (const float*)d_in[0];
  const float* Wqkv  = (const float*)d_in[1];
  const float* bqkv  = (const float*)d_in[2];
  const float* Wo    = (const float*)d_in[3];
  const float* bo    = (const float*)d_in[4];
  const float* Wfc   = (const float*)d_in[5];
  const float* bfc   = (const float*)d_in[6];
  const float* Wproj = (const float*)d_in[7];
  const float* bproj = (const float*)d_in[8];
  float* out = (float*)d_out;

  const int Bn = 2, T = 2048, C = 1024, M = Bn * T;

  // workspace layout (bytes), with reuse:
  //   [WqkvT 6M][WoT 2M][WfcT 8M][WprojT 8M][Xh 8M (reused as x1h)]
  //   [x1 16M][qkv 24M + yh 8M  -> reused as hh 32M]
  char* ws = (char*)d_ws;
  size_t off = 0;
  auto carve = [&](size_t bytes) { char* p = ws + off; off += (bytes + 255) & ~(size_t)255; return p; };
  h16*   WqkvT  = (h16*)carve((size_t)3 * C * C * 2);
  h16*   WoT    = (h16*)carve((size_t)C * C * 2);
  h16*   WfcT   = (h16*)carve((size_t)4 * C * C * 2);
  h16*   WprojT = (h16*)carve((size_t)4 * C * C * 2);
  h16*   Xh     = (h16*)carve((size_t)M * C * 2);       // also x1h
  float* x1     = (float*)carve((size_t)M * C * 4);
  char*  region = carve((size_t)M * 4 * C * 2);         // 32 MB
  h16*   qkv = (h16*)region;                            // 24 MB
  h16*   yh  = (h16*)(region + (size_t)M * 3 * C * 2);  // 8 MB
  h16*   hh  = (h16*)region;                            // reuse after attention+Wo
  h16*   x1h = Xh;

  // 1. casts / weight transposes
  cast_f32_f16<<<(M * C + 255) / 256, 256, 0, stream>>>(x, Xh, M * C);
  transpose_cast<<<dim3(3 * C / 32, C / 32), 256, 0, stream>>>(Wqkv, WqkvT, C, 3 * C);
  transpose_cast<<<dim3(C / 32, C / 32), 256, 0, stream>>>(Wo, WoT, C, C);
  transpose_cast<<<dim3(4 * C / 32, C / 32), 256, 0, stream>>>(Wfc, WfcT, C, 4 * C);
  transpose_cast<<<dim3(C / 32, 4 * C / 32), 256, 0, stream>>>(Wproj, WprojT, 4 * C, C);

  // 2. qkv = x @ Wqkv + bqkv  (f16 out)
  gemm_f16<<<dim3(3 * C / 128, M / 128), 256, 0, stream>>>(
      Xh, WqkvT, bqkv, nullptr, nullptr, qkv, M, 3 * C, C, 0);

  // 3. flash attention -> yh (f16)
  attn_kernel<<<dim3(T / 128, 16, Bn), 256, 0, stream>>>(qkv, yh);

  // 4. x1 = yh @ Wo + bo + x   (f32 + f16 copies)
  gemm_f16<<<dim3(C / 128, M / 128), 256, 0, stream>>>(
      yh, WoT, bo, x, x1, x1h, M, C, C, 0);

  // 5. hh = gelu(x1 @ Wfc + bfc)  (f16)
  gemm_f16<<<dim3(4 * C / 128, M / 128), 256, 0, stream>>>(
      x1h, WfcT, bfc, nullptr, nullptr, hh, M, 4 * C, C, 1);

  // 6. out = hh @ Wproj + bproj + x1  (f32)
  gemm_f16<<<dim3(C / 128, M / 128), 256, 0, stream>>>(
      hh, WprojT, bproj, x1, out, nullptr, M, C, 4 * C, 0);
}
